// Model_7352984010779
// MI455X (gfx1250) — compile-verified, matrix operand order checked by reading, not run on hardware
//
#include <hip/hip_runtime.h>
#include <hip/hip_bf16.h>

// ---------------------------------------------------------------------------
// Problem constants (match reference)
// ---------------------------------------------------------------------------
#define SEQ_LEN 131072
#define IN_DIM  8
#define HID     50
#define NLAY    5
#define HPAD    64   // padded H for LDS rows

typedef __attribute__((ext_vector_type(16))) _Float16 v16h;
typedef __attribute__((ext_vector_type(8)))  float    v8f;

// Native v_rcp_f32 (1 ulp) instead of the IEEE div expansion.
__device__ __forceinline__ float fast_rcp(float v) {
#if __has_builtin(__builtin_amdgcn_rcpf)
    return __builtin_amdgcn_rcpf(v);
#else
    return 1.0f / v;
#endif
}

// Native gfx1250 tanh (v_tanh_f32) if available; else a short native-exp/rcp
// expansion: tanh(x) = 2/(1+exp(-2x)) - 1  ->  v_exp_f32 + v_rcp_f32.
__device__ __forceinline__ float fast_tanh(float v) {
#if __has_builtin(__builtin_amdgcn_tanhf)
    return __builtin_amdgcn_tanhf(v);
#else
    const float t = fast_rcp(1.0f + __expf(-2.0f * v));
    return 2.0f * t - 1.0f;
#endif
}

__device__ __forceinline__ float fast_sigmoid(float v) {
    return fast_rcp(1.0f + __expf(-v));
}

// ---------------------------------------------------------------------------
// Kernel 1: layer-pipelined recurrent scan.
//   One workgroup, 10 waves (2 waves per layer). Wave-pair (2*32=64 lanes)
//   for layer l: lane owns output neuron j (j<50 active). Weights cached in
//   VGPRs. h vectors handed between layers through double-buffered LDS with
//   one workgroup barrier per global step. Global step s: layer l processes
//   timestep t = s - l  (software pipeline, skew 1).
//   Layer 4 streams h4[t][0..49] to global workspace for the WMMA head.
// ---------------------------------------------------------------------------
__global__ __launch_bounds__(NLAY * 64) void rnn_scan_kernel(
    const float* __restrict__ x,      // [L, 8]
    const float* __restrict__ Wih0,   // [50, 8]
    const float* __restrict__ WihR,   // [4, 50, 50]
    const float* __restrict__ Whh,    // [5, 50, 50]
    const float* __restrict__ bih,    // [5, 50]
    const float* __restrict__ bhh,    // [5, 50]
    float* __restrict__ h4)           // [L, 50] workspace out
{
    __shared__ float hbuf[2][NLAY][HPAD];

    const int tid  = threadIdx.x;
    const int wv   = tid >> 5;         // wave id 0..9
    const int lane = tid & 31;
    const int l    = wv >> 1;          // layer 0..4
    const int j    = ((wv & 1) << 5) + lane;   // output neuron 0..63
    const bool jv  = (j < HID);
    const int jj   = jv ? j : 0;       // clamped for safe weight loads

    // Zero both h buffers (h_{-1} = 0 for every layer).
    for (int k = tid; k < 2 * NLAY * HPAD; k += NLAY * 64)
        ((float*)hbuf)[k] = 0.0f;

    // ---- cache this lane's weights in VGPRs ----
    float wih[HID];   // layer 0 uses only wih[0..7] (input projection)
    float whh[HID];
    if (l == 0) {
        #pragma unroll
        for (int i = 0; i < IN_DIM; ++i) wih[i] = Wih0[jj * IN_DIM + i];
    } else {
        #pragma unroll
        for (int i = 0; i < HID; ++i) wih[i] = WihR[((l - 1) * HID + jj) * HID + i];
    }
    #pragma unroll
    for (int i = 0; i < HID; ++i) whh[i] = Whh[(l * HID + jj) * HID + i];
    const float bias = bih[l * HID + jj] + bhh[l * HID + jj];

    __syncthreads();

    int cur = 0;
    const int nsteps = SEQ_LEN + NLAY - 1;
    for (int s = 0; s < nsteps; ++s) {
        const int t = s - l;
        const bool active = (t >= 0) & (t < SEQ_LEN);
        float hnew = 0.0f;

        if (active) {
            float acc = bias;
            if (l == 0) {
                // input projection: x[t][0..7] broadcast to all lanes
                const float4* xr = (const float4*)(x + (size_t)t * IN_DIM);
                float4 xa = xr[0], xb = xr[1];
                acc += xa.x * wih[0] + xa.y * wih[1] + xa.z * wih[2] + xa.w * wih[3]
                     + xb.x * wih[4] + xb.y * wih[5] + xb.z * wih[6] + xb.w * wih[7];
                if (t + 32 < SEQ_LEN)   // stream upcoming x rows into cache
                    __builtin_prefetch(x + (size_t)(t + 32) * IN_DIM, 0, 1);
            } else {
                const float* hin = hbuf[cur][l - 1];   // h_{l-1, t}
                #pragma unroll
                for (int i = 0; i < HID; ++i) acc += hin[i] * wih[i];
            }
            const float* hpr = hbuf[cur][l];           // h_{l, t-1}
            #pragma unroll
            for (int i = 0; i < HID; ++i) acc += hpr[i] * whh[i];
            hnew = fast_tanh(acc);
        }

        // writes go to the other buffer; readers of it were separated by the
        // barrier at the end of the previous step.
        if (active && jv) {
            hbuf[cur ^ 1][l][j] = hnew;
            if (l == NLAY - 1) h4[(size_t)t * HID + j] = hnew;
        }
        __syncthreads();
        cur ^= 1;
    }
}

// ---------------------------------------------------------------------------
// Kernel 2: WMMA head.  out[t] = sigmoid(relu(h4[t]·W1^T + b1)·W2^T + b2)
//   Per wave: one 16-timestep tile.  A = h4 tile (16x64 f16, K padded),
//   B = W1^T tiles (64x16 and 64x4-pad f16).  4x v_wmma_f32_16x16x32_f16
//   per tile, then relu+bias, 20-wide contraction via shfl_xor within
//   16-lane groups, native-rcp sigmoid, store.  EXEC uniform at every WMMA.
// ---------------------------------------------------------------------------

// WMMA 16-bit A layout (wave32): lane m (0..15) row m holds K {0..7,16..23},
// lane m+16 row m holds K {8..15,24..31}. Element e -> k offset:
__device__ __forceinline__ int wmma_k_of(int e, int hi) {
    return (e < 8 ? e : e + 8) + hi * 8;
}

__global__ __launch_bounds__(256) void head_wmma_kernel(
    const float* __restrict__ h4,    // [L, 50]
    const float* __restrict__ W1,    // [20, 50]
    const float* __restrict__ b1,    // [20]
    const float* __restrict__ W2,    // [1, 20]
    const float* __restrict__ b2,    // [1]
    float* __restrict__ out,         // [L]
    int ntiles)
{
    const int lane = threadIdx.x & 31;
    const int lm   = lane & 15;        // row (A) / col (B) inside tile
    const int hi   = lane >> 4;        // which K-half this lane carries
    const int wvG  = blockIdx.x * (blockDim.x >> 5) + (threadIdx.x >> 5);
    const int nwv  = gridDim.x * (blockDim.x >> 5);

    // ---- B tiles (W1^T), built once per wave: [ntile][kstep] ----
    v16h B00, B01, B10, B11;
    #pragma unroll
    for (int e = 0; e < 16; ++e) {
        const int k0 = wmma_k_of(e, hi);        // kstep 0: K 0..31
        const int k1 = k0 + 32;                 // kstep 1: K 32..63
        const int c0 = lm;                      // ntile 0: cols 0..15
        const int c1 = 16 + lm;                 // ntile 1: cols 16..19 (+pad)
        B00[e] = (_Float16)(W1[c0 * HID + k0]);                       // k0 < 50 always
        B01[e] = (_Float16)((k1 < HID) ? W1[c0 * HID + k1] : 0.0f);
        B10[e] = (_Float16)((c1 < 20) ? W1[c1 * HID + k0] : 0.0f);
        B11[e] = (_Float16)((c1 < 20 && k1 < HID) ? W1[c1 * HID + k1] : 0.0f);
    }
    float w2r[20];
    #pragma unroll
    for (int i = 0; i < 20; ++i) w2r[i] = W2[i];
    const float bb2  = b2[0];
    const float b1n0 = b1[lm];                       // col lm (< 16 < 20)
    const float b1n1 = (16 + lm < 20) ? b1[16 + lm] : 0.0f;

    for (int tile = wvG; tile < ntiles; tile += nwv) {
        // ---- A tile: 16 rows of h4, f32 -> f16, K padded to 64 ----
        const float* hrow = h4 + (size_t)(tile * 16 + lm) * HID;
        v16h A0, A1;
        #pragma unroll
        for (int e = 0; e < 16; ++e) {
            const int k0 = wmma_k_of(e, hi);
            const int k1 = k0 + 32;
            A0[e] = (_Float16)(hrow[k0]);
            A1[e] = (_Float16)((k1 < HID) ? hrow[k1] : 0.0f);
        }

        v8f c0 = {}, c1 = {};
        c0 = __builtin_amdgcn_wmma_f32_16x16x32_f16(false, A0, false, B00,
                                                    (short)0, c0, false, false);
        c0 = __builtin_amdgcn_wmma_f32_16x16x32_f16(false, A1, false, B01,
                                                    (short)0, c0, false, false);
        c1 = __builtin_amdgcn_wmma_f32_16x16x32_f16(false, A0, false, B10,
                                                    (short)0, c1, false, false);
        c1 = __builtin_amdgcn_wmma_f32_16x16x32_f16(false, A1, false, B11,
                                                    (short)0, c1, false, false);

        // D layout: VGPR r -> (row r, col lm) for lanes 0-15,
        //                     (row r+8, col lm) for lanes 16-31.
        #pragma unroll
        for (int r = 0; r < 8; ++r) {
            float z0 = c0[r] + b1n0;  z0 = z0 > 0.0f ? z0 : 0.0f;
            float p  = z0 * w2r[lm];
            if (lm < 4) {
                float z1 = c1[r] + b1n1;  z1 = z1 > 0.0f ? z1 : 0.0f;
                p += z1 * w2r[16 + lm];
            }
            // reduce the 20-col contraction across the 16-lane group
            p += __shfl_xor(p, 8, 16);
            p += __shfl_xor(p, 4, 16);
            p += __shfl_xor(p, 2, 16);
            p += __shfl_xor(p, 1, 16);
            if (lm == 0) {
                out[tile * 16 + r + hi * 8] = fast_sigmoid(p + bb2);
            }
        }
    }
}

// ---------------------------------------------------------------------------
// Launch
// ---------------------------------------------------------------------------
extern "C" void kernel_launch(void* const* d_in, const int* in_sizes, int n_in,
                              void* d_out, int out_size, void* d_ws, size_t ws_size,
                              hipStream_t stream) {
    const float* x    = (const float*)d_in[0];
    const float* Wih0 = (const float*)d_in[1];
    const float* WihR = (const float*)d_in[2];
    const float* Whh  = (const float*)d_in[3];
    const float* bih  = (const float*)d_in[4];
    const float* bhh  = (const float*)d_in[5];
    const float* W1   = (const float*)d_in[6];
    const float* b1   = (const float*)d_in[7];
    const float* W2   = (const float*)d_in[8];
    const float* b2   = (const float*)d_in[9];
    float* out = (float*)d_out;
    float* h4  = (float*)d_ws;            // SEQ_LEN*HID floats = 26.2 MB

    rnn_scan_kernel<<<1, NLAY * 64, 0, stream>>>(x, Wih0, WihR, Whh, bih, bhh, h4);

    const int ntiles = SEQ_LEN / 16;      // 8192
    head_wmma_kernel<<<256, 256, 0, stream>>>(h4, W1, b1, W2, b2, out, ntiles);
}
